// FovConv2dCont_49486613184965
// MI455X (gfx1250) — matrix operand order repaired and checked by reading, not compiled, not run on hardware
//
#include <hip/hip_runtime.h>

// ---------------------------------------------------------------------------
// FovConv2dCont for MI455X (gfx1250, wave32, WMMA)
//   stage 1: per-pixel Gaussian (sigma ~ distance to FOA), 7x7, normalized
//   stage 2: 5x5 conv 3->64 channels == implicit GEMM M=B*H*W, N=64, K=75
// Stage 2: V_WMMA_F32_16X16X32_F16, K padded 75->96 (3 WMMAs per 16x16 tile),
// 32-pixel M segment per block (2 M-tiles x 4 N-tiles), B fragments reused.
// All A-fragment LDS gathers use compile-time offsets (template over K-base
// and wave-half) -> ds_load_u16 with immediate offsets, no runtime div/mod.
// Stage 1 uses ONE v_exp_f32 per pixel: exp(-d2*a) = t^d2, d2 in a 10-value
// integer set -> multiply chain instead of 49 transcendentals.
// ---------------------------------------------------------------------------

typedef __attribute__((ext_vector_type(16))) _Float16 v16h;
typedef __attribute__((ext_vector_type(8)))  float    v8f;

#define BATCH 8
#define CH_IN 3
#define CH_OUT 64
#define H 224
#define W 224
#define KG 7                      // gaussian kernel
#define KC 5                      // conv kernel
#define KTOT (CH_IN * KC * KC)    // 75
#define MSEG 32                   // pixels per block (two 16-wide M tiles)
#define WTILES (W / MSEG)         // 7
#define TILE_COLS (MSEG + 4)      // 36 halo columns
#define SMEM_ELEMS (CH_IN * 5 * TILE_COLS)   // 540

// ---------------------------------------------------------------------------
// Kernel 0: pack weights (64,3,5,5) f32 -> B fragments, f16.
// B-matrix 32x16 (KxN), 16-bit layout (ISA 7.12.2 / SWMMAC B pattern):
//   lanes 0-15 : N = lane,    V0..3 -> K 0..7,  V4..7 -> K 16..23
//   lanes 16-31: N = lane-16, V0..3 -> K 8..15, V4..7 -> K 24..31
// bpack layout: [ntile(4)][kstep(3)][lane(32)][16 halfs]  (32B per lane)
// ---------------------------------------------------------------------------
__global__ void fov_pack_weights(const float* __restrict__ wgt,
                                 _Float16* __restrict__ bpack) {
    const int ntile = blockIdx.x / 3;
    const int kstep = blockIdx.x % 3;
    const int lane  = threadIdx.x;          // 0..31
    const int n     = ntile * 16 + (lane & 15);
    const bool lo   = lane < 16;
    const int base  = ((ntile * 3 + kstep) * 32 + lane) * 16;  // in halfs

#pragma unroll
    for (int v = 0; v < 8; ++v) {
        const int kl = (v < 4 ? 0 : 8) + (lo ? 0 : 8) + 2 * v;
#pragma unroll
        for (int s = 0; s < 2; ++s) {
            const int k = kstep * 32 + kl + s;
            _Float16 val = (_Float16)0.0f;
            if (k < KTOT) {
                // k = c*25 + ki*5 + kj matches (64,3,5,5) row-major flat index
                val = (_Float16)wgt[n * KTOT + k];
            }
            bpack[base + 2 * v + s] = val;
        }
    }
}

// ---------------------------------------------------------------------------
// Kernel 1: Gaussian FOA modulation, f32 in -> f16 out (feeds WMMA A-matrix).
// One thread per (b, pixel). exp(-d2*a) = t^d2 with t = exp(-a); d2 takes
// only {0,1,2,4,5,8,9,10,13,18} on the 7x7 grid -> 1 exp + 8 muls.
// Normalization (1/sum) folded into the 10 unique weights once.
// ---------------------------------------------------------------------------
__global__ __launch_bounds__(256)
void fov_modulate(const float* __restrict__ x,
                  const int* __restrict__ foa,
                  _Float16* __restrict__ mod) {
    const int idx = blockIdx.x * 256 + threadIdx.x;
    if (idx >= BATCH * H * W) return;
    const int b = idx / (H * W);
    const int p = idx % (H * W);
    const int h = p / W;
    const int w = p % W;

    const float fr = (float)foa[b * 2 + 0];
    const float fc = (float)foa[b * 2 + 1];
    const float dr = (float)h - fr;
    const float dc = (float)w - fc;
    // 1/sqrt(224^2 + 224^2)
    const float dist = sqrtf(dr * dr + dc * dc) * 0.0031567583f;
    const float sigma = (1.0f - dist) * 0.01f + dist * 1.0f;
    const float a = 1.0f / (2.0f * sigma * sigma);

    // t^d2 power chain (single transcendental)
    const float t1  = __expf(-a);
    const float t2  = t1 * t1;
    const float t4  = t2 * t2;
    const float t5  = t4 * t1;
    const float t8  = t4 * t4;
    const float t9  = t8 * t1;
    const float t10 = t5 * t5;
    const float t13 = t8 * t5;
    const float t18 = t9 * t9;

    float tp[19];
    tp[0] = 1.0f; tp[1] = t1;  tp[2] = t2;  tp[4] = t4;  tp[5] = t5;
    tp[8] = t8;   tp[9] = t9;  tp[10] = t10; tp[13] = t13; tp[18] = t18;

    // sum over 7x7 grid via multiplicities of each d2 value
    const float s = 1.0f * tp[0] + 4.0f * t1 + 4.0f * t2 + 4.0f * t4 +
                    8.0f * t5 + 4.0f * t8 + 4.0f * t9 + 8.0f * t10 +
                    8.0f * t13 + 4.0f * t18;
    const float inv = 1.0f / s;

    // normalized weights (only the 10 used entries matter)
    tp[0] *= inv; tp[1] = t1 * inv;  tp[2] = t2 * inv;  tp[4] = t4 * inv;
    tp[5] = t5 * inv;  tp[8] = t8 * inv;  tp[9] = t9 * inv;
    tp[10] = t10 * inv; tp[13] = t13 * inv; tp[18] = t18 * inv;

#pragma unroll
    for (int c = 0; c < CH_IN; ++c) {
        const float* xc = x + ((b * CH_IN + c) * H) * W;
        float acc = 0.0f;
#pragma unroll
        for (int i = 0; i < KG; ++i) {
            const int sh = h + i - 3;
            if (sh < 0 || sh >= H) continue;
#pragma unroll
            for (int j = 0; j < KG; ++j) {
                const int sw = w + j - 3;
                if (sw < 0 || sw >= W) continue;
                const int d2 = (i - 3) * (i - 3) + (j - 3) * (j - 3);  // const
                acc += tp[d2] * xc[sh * W + sw];
            }
        }
        mod[((b * CH_IN + c) * H + h) * W + w] = (_Float16)acc;
    }
}

// ---------------------------------------------------------------------------
// A-fragment builder, 16-bit A-matrix 16x32 layout (ISA 7.12.2):
//   lanes 0-15 (M=lane):     V0..3 -> K 0..7,   V4..7 -> K 16..23
//   lanes 16-31 (M=lane-16): V0..3 -> K 8..15,  V4..7 -> K 24..31
// KBASE and LO are compile-time -> every k, (c,ki,kj) and smem offset folds
// to a constant; gathers become ds_load_u16 [2*col + imm]; k>=75 -> literal 0.
// ---------------------------------------------------------------------------
template <int KBASE, bool LO>
__device__ __forceinline__ v16h build_A(const _Float16* smem, int col) {
    union { v16h v; _Float16 e[16]; } A;
#pragma unroll
    for (int vv = 0; vv < 8; ++vv) {
#pragma unroll
        for (int s = 0; s < 2; ++s) {
            const int kl = (vv < 4) ? ((LO ? 0 : 8) + 2 * vv)
                                    : ((LO ? 16 : 24) + 2 * (vv - 4));
            const int k = KBASE + kl + s;         // compile-time constant
            if (k < KTOT) {
                const int c  = k / 25;
                const int tt = k % 25;
                const int ki = tt / 5;
                const int kj = tt % 5;
                A.e[2 * vv + s] = smem[c * (5 * TILE_COLS) + ki * TILE_COLS + kj + col];
            } else {
                A.e[2 * vv + s] = (_Float16)0.0f;
            }
        }
    }
    return A.v;
}

// ---------------------------------------------------------------------------
// Kernel 2: 5x5 conv as implicit GEMM via WMMA f32_16x16x32_f16.
// Block = 128 threads (4 waves). Block covers a 32-pixel row segment
// (fixed b,h; w in [w0,w0+32)) for all 64 output channels.
// Wave i owns N-tile i and computes both M-tiles, reusing B fragments.
// ---------------------------------------------------------------------------
__global__ __launch_bounds__(128)
void fov_conv5(const _Float16* __restrict__ mod,
               const _Float16* __restrict__ bpack,
               float* __restrict__ out) {
    __shared__ _Float16 smem[SMEM_ELEMS];   // [c(3)][row(5)][col(36)]

    const int bid = blockIdx.x;
    const int wt = bid % WTILES;
    const int t1 = bid / WTILES;
    const int h  = t1 % H;
    const int b  = t1 / H;
    const int w0 = wt * MSEG;

    // ---- stage halo tile into LDS (zero padding at borders) ----
    for (int i = threadIdx.x; i < SMEM_ELEMS; i += 128) {
        const int c  = i / (5 * TILE_COLS);
        const int rr = i % (5 * TILE_COLS);
        const int r  = rr / TILE_COLS;
        const int cc = rr % TILE_COLS;
        const int sh = h + r - 2;
        const int sw = w0 + cc - 2;
        _Float16 v = (_Float16)0.0f;
        if (sh >= 0 && sh < H && sw >= 0 && sw < W)
            v = mod[((b * CH_IN + c) * H + sh) * W + sw];
        smem[i] = v;
    }
    __syncthreads();

    const int wave = threadIdx.x >> 5;    // 0..3 == N-tile
    const int lane = threadIdx.x & 31;
    const int m    = lane & 15;

    // ---- gather A fragments: one branch per wave-half, constant offsets ----
    v16h A0, A1, A2;      // M-tile 0 (pixels w0..w0+15), ksteps 0..2
    v16h A3, A4, A5;      // M-tile 1 (pixels w0+16..w0+31)
    if (lane < 16) {
        A0 = build_A< 0, true>(smem, m);
        A1 = build_A<32, true>(smem, m);
        A2 = build_A<64, true>(smem, m);
        A3 = build_A< 0, true>(smem, m + 16);
        A4 = build_A<32, true>(smem, m + 16);
        A5 = build_A<64, true>(smem, m + 16);
    } else {
        A0 = build_A< 0, false>(smem, m);
        A1 = build_A<32, false>(smem, m);
        A2 = build_A<64, false>(smem, m);
        A3 = build_A< 0, false>(smem, m + 16);
        A4 = build_A<32, false>(smem, m + 16);
        A5 = build_A<64, false>(smem, m + 16);
    }

    // ---- B fragments: prepacked, 32B per lane (two b128 loads each) ----
    const _Float16* bp = bpack + (wave * 3 * 32 + lane) * 16;
    const v16h B0 = *(const v16h*)(bp + 0 * 32 * 16);
    const v16h B1 = *(const v16h*)(bp + 1 * 32 * 16);
    const v16h B2 = *(const v16h*)(bp + 2 * 32 * 16);

    // ---- 6 WMMAs: two M-tiles x 3 K-steps, B reused across M-tiles ----
    v8f acc0 = {}, acc1 = {};
    acc0 = __builtin_amdgcn_wmma_f32_16x16x32_f16(false, A0, false, B0, (short)0, acc0, false, false);
    acc1 = __builtin_amdgcn_wmma_f32_16x16x32_f16(false, A3, false, B0, (short)0, acc1, false, false);
    acc0 = __builtin_amdgcn_wmma_f32_16x16x32_f16(false, A1, false, B1, (short)0, acc0, false, false);
    acc1 = __builtin_amdgcn_wmma_f32_16x16x32_f16(false, A4, false, B1, (short)0, acc1, false, false);
    acc0 = __builtin_amdgcn_wmma_f32_16x16x32_f16(false, A2, false, B2, (short)0, acc0, false, false);
    acc1 = __builtin_amdgcn_wmma_f32_16x16x32_f16(false, A5, false, B2, (short)0, acc1, false, false);

    // ---- store D (ISA C/D layout: VGPR r -> M = base_m + r, N = lane&15) ----
    // per-lane per-tile: 8 consecutive floats along W -> two aligned b128s
    const int n = wave * 16 + m;
    const int base_m = (lane < 16) ? 0 : 8;
    float* o = out + (((b * CH_OUT + n) * H + h) * W + w0 + base_m);
    *(float4*)(o + 0)      = *((const float4*)&acc0 + 0);
    *(float4*)(o + 4)      = *((const float4*)&acc0 + 1);
    *(float4*)(o + 16)     = *((const float4*)&acc1 + 0);
    *(float4*)(o + 16 + 4) = *((const float4*)&acc1 + 1);
}

// ---------------------------------------------------------------------------
extern "C" void kernel_launch(void* const* d_in, const int* in_sizes, int n_in,
                              void* d_out, int out_size, void* d_ws, size_t ws_size,
                              hipStream_t stream) {
    const float* x   = (const float*)d_in[0];   // (8,3,224,224) f32
    const int*   foa = (const int*)d_in[1];     // (8,2) i32
    const float* wgt = (const float*)d_in[2];   // (64,3,5,5) f32
    float* out = (float*)d_out;                 // (8,64,224,224) f32

    // workspace: modulated f16 tensor, then packed B fragments (32B aligned)
    _Float16* mod   = (_Float16*)d_ws;
    _Float16* bpack = mod + (size_t)BATCH * CH_IN * H * W;  // 2,408,448 B offset

    fov_pack_weights<<<12, 32, 0, stream>>>(wgt, bpack);

    const int npix = BATCH * H * W;
    fov_modulate<<<(npix + 255) / 256, 256, 0, stream>>>(x, foa, mod);

    fov_conv5<<<BATCH * H * WTILES, 128, 0, stream>>>(mod, bpack, out);
}
